// completeNet_44057774522894
// MI455X (gfx1250) — matrix operand order repaired and checked by reading, not compiled, not run on hardware
//
#include <hip/hip_runtime.h>
#include <hip/hip_bf16.h>

// ---------------------------------------------------------------------------
// completeNet forward for MI455X (gfx1250, wave32, WMMA f32<-f16 16x16x32).
// All GEMM-shaped math (conv2-as-GEMM, fc, app, msg, upd) runs on
// v_wmma_f32_16x16x32_f16 with f16 inputs / f32 accumulation; scalar-sensitive
// parts (conv1, tiny MLPs, cosine, Sinkhorn) stay fp32 VALU.
// Weights for the WMMA GEMMs are pre-packed into fragment-major layout
// ([kt][nt][lane][16 halfs]) so every B fragment is two global_load_b128.
// Input pointer order assumes setup_inputs() dict flattened with the params
// dict expanded in insertion order (indices 5..34), edge_index at 35.
// ---------------------------------------------------------------------------

typedef __attribute__((ext_vector_type(16))) _Float16 v16h;
typedef __attribute__((ext_vector_type(8)))  float    v8f;

#define BB   8
#define TT   64
#define DDIM 64
#define NPG  128
#define NNODE 1024
#define EFWD 32768
#define ETOT 65536
#define LAMV 5.0f

#define WMMA_F16(a, b, c) \
  __builtin_amdgcn_wmma_f32_16x16x32_f16(false, (a), false, (b), (short)0, (c), false, false)

// A fragment: 16x32 f16, row-major source (lda elems). Contiguous-K layout (same as B):
// lane m=l&15, half=l>>4; VGPR v: K = half*16 + v*2 (+0,+1).
// Per lane this is two contiguous 16-byte runs -> merges to 2x b128.
static __device__ inline v16h wmma_load_a(const _Float16* A, int lda, int row0, int k0, int lane) {
  const _Float16* p = A + (size_t)(row0 + (lane & 15)) * lda;
  int h = lane >> 4;
  v16h a;
#pragma unroll
  for (int v = 0; v < 8; ++v) {
    int kk = k0 + h * 16 + v * 2;
    a[2 * v]     = p[kk];
    a[2 * v + 1] = p[kk + 1];
  }
  return a;
}

// B fragment from fragment-major packed weights: packed[((kt*NT+nt)*32+lane)*16+j]
// holds W[(kt*32 + (lane>>4)*16 + j) * N + nt*16 + (lane&15)].  32 contiguous
// bytes per lane -> exactly 2x global_load_b128.
static __device__ inline v16h wmma_load_b_packed(const _Float16* Bp, int NT, int kt, int nt, int lane) {
  const _Float16* p = Bp + ((size_t)(kt * NT + nt) * 32 + lane) * 16;
  v16h b;
#pragma unroll
  for (int j = 0; j < 16; ++j) b[j] = p[j];
  return b;
}

// ---------------------------------------------------------------------------
// Weight preparation.
// ---------------------------------------------------------------------------
// Pack a K x N (row-major, f32) weight matrix into WMMA-fragment-major f16.
__global__ void k_prep_packB(const float* __restrict__ W, _Float16* __restrict__ dst,
                             int K, int N) {
  int i = blockIdx.x * 256 + threadIdx.x;
  if (i < K * N) {
    int j    = i & 15;
    int lane = (i >> 4) & 31;
    int rest = i >> 9;
    int NT   = N >> 4;
    int nt   = rest % NT;
    int kt   = rest / NT;
    int k = kt * 32 + (lane >> 4) * 16 + j;
    int n = nt * 16 + (lane & 15);
    dst[i] = (_Float16)W[(size_t)k * N + n];
  }
}

// conv2_w (64,32,3,3) -> f16 [oc][K] with K = (ky*3+kx)*32 + ic (channel-minor
// per filter tap) so each K=32 WMMA step is one (ky,kx) tap over 32 channels.
__global__ void k_prep_c2(const float* __restrict__ w, _Float16* __restrict__ dst) {
  int i = blockIdx.x * 256 + threadIdx.x;
  if (i < 64 * 288) {
    int oc = i / 288, k = i - oc * 288;
    int ks = k >> 5, ic = k & 31;
    int ky = ks / 3, kx = ks - ky * 3;
    dst[i] = (_Float16)w[(size_t)(oc * 32 + ic) * 9 + ky * 3 + kx];
  }
}

// ---------------------------------------------------------------------------
// conv1: 3->32, stride 2, SAME (pad_lo=0, pad_hi=1), ReLU.
// One block per image; output f16 layout [sp(32x32)][ch(32)] (channel-minor)
// to feed conv2 B fragments contiguously.
// ---------------------------------------------------------------------------
__global__ void __launch_bounds__(256) k_conv1(const float* __restrict__ x,
                                               const float* __restrict__ c1w,
                                               const float* __restrict__ c1b,
                                               _Float16* __restrict__ c1g) {
  __shared__ _Float16 imgL[3 * 64 * 64];  // 24KB
  __shared__ float w1L[864];
  __shared__ float b1L[32];
  int tid = threadIdx.x, img = blockIdx.x;
  for (int i = tid; i < 3 * 64 * 64; i += 256) imgL[i] = (_Float16)x[(size_t)img * 12288 + i];
  for (int i = tid; i < 864; i += 256) w1L[i] = c1w[i];
  if (tid < 32) b1L[tid] = c1b[tid];
  __syncthreads();
  for (int idx = tid; idx < 32 * 32 * 32; idx += 256) {
    int oc = idx & 31;
    int sp = idx >> 5;
    int oy = sp >> 5, ox = sp & 31;
    float acc = b1L[oc];
    const float* wp = &w1L[oc * 27];
#pragma unroll
    for (int ic = 0; ic < 3; ++ic)
#pragma unroll
      for (int ky = 0; ky < 3; ++ky) {
        int iy = 2 * oy + ky;
        if (iy >= 64) continue;
#pragma unroll
        for (int kx = 0; kx < 3; ++kx) {
          int ix = 2 * ox + kx;
          if (ix >= 64) continue;
          acc += (float)imgL[ic * 4096 + iy * 64 + ix] * wp[ic * 9 + ky * 3 + kx];
        }
      }
    c1g[(size_t)img * 32768 + sp * 32 + oc] = (_Float16)fmaxf(acc, 0.0f);
  }
}

// ---------------------------------------------------------------------------
// conv2 (32->64, s2, SAME) + ReLU + 16x16 mean pool, as implicit GEMM:
// M=64 out-ch, N=256 spatial, K=288 = 9 taps x 32 ch.  One block per image,
// 8 waves x 8 tiles.  B fragments read straight from L2-resident c1g.
// ---------------------------------------------------------------------------
__global__ void __launch_bounds__(256) k_conv2(const _Float16* __restrict__ c1g,
                                               const _Float16* __restrict__ c2wr16,
                                               const float* __restrict__ c2b,
                                               _Float16* __restrict__ pool16) {
  __shared__ float b2L[64];
  __shared__ float poolL[64];
  int tid = threadIdx.x, img = blockIdx.x;
  if (tid < 64) { b2L[tid] = c2b[tid]; poolL[tid] = 0.0f; }
  __syncthreads();
  const _Float16* c1 = c1g + (size_t)img * 32768;
  int wave = tid >> 5, lane = tid & 31;
  int nloc = lane & 15, hloc = lane >> 4;
  for (int t = wave; t < 64; t += 8) {
    int mt = t >> 4, nt = t & 15;
    int n = nt * 16 + nloc;
    int oy = n >> 4, ox = n & 15;
    v8f c = {};
#pragma unroll
    for (int ks = 0; ks < 9; ++ks) {
      int ky = ks / 3, kx = ks - ky * 3;
      int iy = 2 * oy + ky, ix = 2 * ox + kx;
      v16h a = wmma_load_a(c2wr16, 288, mt * 16, ks * 32, lane);
      v16h b;
      if (iy < 32 && ix < 32) {
        const _Float16* p = c1 + (iy * 32 + ix) * 32 + hloc * 16;  // contiguous 16 ch
#pragma unroll
        for (int v = 0; v < 16; ++v) b[v] = p[v];
      } else {
#pragma unroll
        for (int v = 0; v < 16; ++v) b[v] = (_Float16)0.0f;       // SAME zero pad
      }
      c = WMMA_F16(a, b, c);
    }
#pragma unroll
    for (int r = 0; r < 8; ++r) {
      int oc = mt * 16 + hloc * 8 + r;
      atomicAdd(&poolL[oc], fmaxf(c[r] + b2L[oc], 0.0f) * (1.0f / 256.0f));
    }
  }
  __syncthreads();
  if (tid < 64) pool16[img * 64 + tid] = (_Float16)poolL[tid];
}

// ---------------------------------------------------------------------------
// fc: h = relu(pool @ fc_w + b), 1024 x 64 x 128, 64 rows per block.
// ---------------------------------------------------------------------------
__global__ void __launch_bounds__(256) k_fc(const _Float16* __restrict__ pool16,
                                            const _Float16* __restrict__ fcw16,
                                            const float* __restrict__ fcb,
                                            _Float16* __restrict__ h16) {
  __shared__ _Float16 As[64 * 64];  // 8KB
  int tid = threadIdx.x;
  int row0 = blockIdx.x * 64;
  for (int i = tid; i < 64 * 64; i += 256) As[i] = pool16[(size_t)row0 * 64 + i];
  __syncthreads();
  int wave = tid >> 5, lane = tid & 31;
  for (int t = wave; t < 32; t += 8) {
    int mt = t >> 3, nt = t & 7;
    v8f c = {};
#pragma unroll
    for (int ks = 0; ks < 2; ++ks) {
      v16h a = wmma_load_a(As, 64, mt * 16, ks * 32, lane);
      v16h b = wmma_load_b_packed(fcw16, 8, ks, nt, lane);
      c = WMMA_F16(a, b, c);
    }
    int n = nt * 16 + (lane & 15);
    int mh = (lane >> 4) * 8;
    float bn = fcb[n];
#pragma unroll
    for (int r = 0; r < 8; ++r)
      h16[(size_t)(row0 + mt * 16 + mh + r) * 128 + n] = (_Float16)fmaxf(c[r] + bn, 0.0f);
  }
}

// ---------------------------------------------------------------------------
// app MLP layer 1 as GEMM [64 edges x 256] x [256 x 64] + relu, then 64->1
// reduce for x1.  64 edges per block; A = [h[src] || h[dst]] staged in LDS.
// ---------------------------------------------------------------------------
__global__ void __launch_bounds__(256) k_app(const _Float16* __restrict__ h16,
                                             const _Float16* __restrict__ appw16,
                                             const float* __restrict__ ab1,
                                             const float* __restrict__ aw2,
                                             const float* __restrict__ ab2,
                                             const int* __restrict__ ei,
                                             float* __restrict__ x1) {
  __shared__ _Float16 As[64 * 256];  // 32KB
  __shared__ float hid[64 * 64];     // 16KB
  __shared__ int srcL[64], dstL[64];
  __shared__ float w2L[64];
  int tid = threadIdx.x;
  int e0 = blockIdx.x * 64;
  if (tid < 64) { srcL[tid] = ei[e0 + tid]; dstL[tid] = ei[ETOT + e0 + tid]; w2L[tid] = aw2[tid]; }
  __syncthreads();
  for (int i = tid; i < 64 * 256; i += 256) {
    int r = i >> 8, cc = i & 255;
    As[i] = (cc < 128) ? h16[(size_t)srcL[r] * 128 + cc] : h16[(size_t)dstL[r] * 128 + (cc - 128)];
  }
  __syncthreads();
  int wave = tid >> 5, lane = tid & 31;
  for (int t = wave; t < 16; t += 8) {
    int mt = t >> 2, nt = t & 3;
    v8f c = {};
#pragma unroll
    for (int ks = 0; ks < 8; ++ks) {
      v16h a = wmma_load_a(As, 256, mt * 16, ks * 32, lane);
      v16h b = wmma_load_b_packed(appw16, 4, ks, nt, lane);
      c = WMMA_F16(a, b, c);
    }
    int n = nt * 16 + (lane & 15);
    int mh = (lane >> 4) * 8;
    float bn = ab1[n];
#pragma unroll
    for (int r = 0; r < 8; ++r)
      hid[(mt * 16 + mh + r) * 64 + n] = fmaxf(c[r] + bn, 0.0f);
  }
  __syncthreads();
  if (tid < 64) {
    float acc = ab2[0];
    for (int j = 0; j < 64; ++j) acc += hid[tid * 64 + j] * w2L[j];
    x1[e0 + tid] = acc;
  }
}

// ---------------------------------------------------------------------------
// Per-edge small MLPs: geo (8->32->1) -> x2, aff([x1,x2] 2->32->16) -> e_emb,
// pos (8->32->16) -> p_emb.  ef16 = [e_emb || p_emb] (E x 32, f16).
// ---------------------------------------------------------------------------
__global__ void __launch_bounds__(256) k_edgefeat(
    const float* __restrict__ coords, const float* __restrict__ pea,
    const float* __restrict__ x1,
    const float* __restrict__ gw1, const float* __restrict__ gb1,
    const float* __restrict__ gw2, const float* __restrict__ gb2,
    const float* __restrict__ fw1, const float* __restrict__ fb1,
    const float* __restrict__ fw2, const float* __restrict__ fb2,
    const float* __restrict__ pw1, const float* __restrict__ pb1,
    const float* __restrict__ pw2, const float* __restrict__ pb2,
    const int* __restrict__ ei, _Float16* __restrict__ ef16) {
  __shared__ float sgw1[256], sgb1[32], sgw2[32];
  __shared__ float sfw1[64], sfb1[32], sfw2[512], sfb2[16];
  __shared__ float spw1[256], spb1[32], spw2[512], spb2[16];
  int tid = threadIdx.x;
  sgw1[tid] = gw1[tid];
  spw1[tid] = pw1[tid];
  if (tid < 32) { sgb1[tid] = gb1[tid]; sgw2[tid] = gw2[tid]; sfb1[tid] = fb1[tid]; spb1[tid] = pb1[tid]; }
  if (tid < 64) sfw1[tid] = fw1[tid];
  for (int i = tid; i < 512; i += 256) { sfw2[i] = fw2[i]; spw2[i] = pw2[i]; }
  if (tid < 16) { sfb2[tid] = fb2[tid]; spb2[tid] = pb2[tid]; }
  __syncthreads();

  int e = blockIdx.x * 256 + tid;
  int s = ei[e], d = ei[ETOT + e];
  float in8[8];
#pragma unroll
  for (int i = 0; i < 4; ++i) { in8[i] = coords[s * 4 + i]; in8[4 + i] = coords[d * 4 + i]; }
  float x2 = gb2[0];
  for (int j = 0; j < 32; ++j) {
    float a = sgb1[j];
#pragma unroll
    for (int i = 0; i < 8; ++i) a += in8[i] * sgw1[i * 32 + j];
    x2 += fmaxf(a, 0.0f) * sgw2[j];
  }
  float v1 = x1[e];
  float hid[32];
  for (int j = 0; j < 32; ++j)
    hid[j] = fmaxf(sfb1[j] + v1 * sfw1[j] + x2 * sfw1[32 + j], 0.0f);
  for (int o = 0; o < 16; ++o) {
    float a = sfb2[o];
    for (int j = 0; j < 32; ++j) a += hid[j] * sfw2[j * 16 + o];
    ef16[(size_t)e * 32 + o] = (_Float16)a;            // e_emb
  }
  float pin[8];
#pragma unroll
  for (int i = 0; i < 8; ++i) pin[i] = pea[(size_t)e * 8 + i];
  for (int j = 0; j < 32; ++j) {
    float a = spb1[j];
#pragma unroll
    for (int i = 0; i < 8; ++i) a += pin[i] * spw1[i * 32 + j];
    hid[j] = fmaxf(a, 0.0f);
  }
  for (int o = 0; o < 16; ++o) {
    float a = spb2[o];
    for (int j = 0; j < 32; ++j) a += hid[j] * spw2[j * 16 + o];
    ef16[(size_t)e * 32 + 16 + o] = (_Float16)a;       // p_emb
  }
}

// ---------------------------------------------------------------------------
// msg = relu([h[src] || ef] @ msg_w + b): [64 edges x 160] x [160 x 128].
// ---------------------------------------------------------------------------
__global__ void __launch_bounds__(256) k_msg(const _Float16* __restrict__ h16,
                                             const _Float16* __restrict__ ef16,
                                             const _Float16* __restrict__ msgw16,
                                             const float* __restrict__ msgb,
                                             const int* __restrict__ ei,
                                             float* __restrict__ msgf) {
  __shared__ _Float16 As[64 * 160];  // 20KB
  __shared__ int srcL[64];
  int tid = threadIdx.x;
  int e0 = blockIdx.x * 64;
  if (tid < 64) srcL[tid] = ei[e0 + tid];
  __syncthreads();
  for (int i = tid; i < 64 * 160; i += 256) {
    int r = i / 160, cc = i - r * 160;
    As[i] = (cc < 128) ? h16[(size_t)srcL[r] * 128 + cc]
                       : ef16[(size_t)(e0 + r) * 32 + (cc - 128)];
  }
  __syncthreads();
  int wave = tid >> 5, lane = tid & 31;
  for (int t = wave; t < 32; t += 8) {
    int mt = t >> 3, nt = t & 7;
    v8f c = {};
#pragma unroll
    for (int ks = 0; ks < 5; ++ks) {
      v16h a = wmma_load_a(As, 160, mt * 16, ks * 32, lane);
      v16h b = wmma_load_b_packed(msgw16, 8, ks, nt, lane);
      c = WMMA_F16(a, b, c);
    }
    int n = nt * 16 + (lane & 15);
    int mh = (lane >> 4) * 8;
    float bn = msgb[n];
#pragma unroll
    for (int r = 0; r < 8; ++r)
      msgf[(size_t)(e0 + mt * 16 + mh + r) * 128 + n] = fmaxf(c[r] + bn, 0.0f);
  }
}

// ---------------------------------------------------------------------------
// Deterministic segment-sum exploiting the static bipartite edge layout:
// det node (g,d) sums edges g*4096 + t*64 + d; track node (g,t) sums
// EFWD + g*4096 + t*64 + d.  No float atomics -> bitwise deterministic.
// ---------------------------------------------------------------------------
__global__ void __launch_bounds__(256) k_agg(const float* __restrict__ msgf,
                                             _Float16* __restrict__ agg16) {
  int i = blockIdx.x * 256 + threadIdx.x;  // [0, 1024*128)
  int n = i >> 7, cc = i & 127;
  int g = n >> 7, idx = n & 127;
  float acc = 0.0f;
  if (idx >= TT) {
    size_t base = (size_t)(g * 4096 + (idx - TT)) * 128 + cc;
    for (int t = 0; t < 64; ++t) acc += msgf[base + (size_t)t * 64 * 128];
  } else {
    size_t base = (size_t)(EFWD + g * 4096 + idx * 64) * 128 + cc;
    for (int d = 0; d < 64; ++d) acc += msgf[base + (size_t)d * 128];
  }
  agg16[i] = (_Float16)acc;
}

// ---------------------------------------------------------------------------
// h2 = relu([h || agg] @ upd_w + b): 1024 x 256 x 128.
// ---------------------------------------------------------------------------
__global__ void __launch_bounds__(256) k_upd(const _Float16* __restrict__ h16,
                                             const _Float16* __restrict__ agg16,
                                             const _Float16* __restrict__ updw16,
                                             const float* __restrict__ updb,
                                             float* __restrict__ h2) {
  __shared__ _Float16 As[64 * 256];  // 32KB
  int tid = threadIdx.x;
  int row0 = blockIdx.x * 64;
  for (int i = tid; i < 64 * 256; i += 256) {
    int r = i >> 8, cc = i & 255;
    As[i] = (cc < 128) ? h16[(size_t)(row0 + r) * 128 + cc]
                       : agg16[(size_t)(row0 + r) * 128 + (cc - 128)];
  }
  __syncthreads();
  int wave = tid >> 5, lane = tid & 31;
  for (int t = wave; t < 32; t += 8) {
    int mt = t >> 3, nt = t & 7;
    v8f c = {};
#pragma unroll
    for (int ks = 0; ks < 8; ++ks) {
      v16h a = wmma_load_a(As, 256, mt * 16, ks * 32, lane);
      v16h b = wmma_load_b_packed(updw16, 8, ks, nt, lane);
      c = WMMA_F16(a, b, c);
    }
    int n = nt * 16 + (lane & 15);
    int mh = (lane >> 4) * 8;
    float bn = updb[n];
#pragma unroll
    for (int r = 0; r < 8; ++r)
      h2[(size_t)(row0 + mt * 16 + mh + r) * 128 + n] = fmaxf(c[r] + bn, 0.0f);
  }
}

// ---------------------------------------------------------------------------
// Per-forward-edge cosine + IoU + fin MLP -> M[g,row,col] = exp(score*LAM).
// ---------------------------------------------------------------------------
__global__ void __launch_bounds__(256) k_score(const float* __restrict__ h2,
                                               const float* __restrict__ co,
                                               const float* __restrict__ fw1,
                                               const float* __restrict__ fb1,
                                               const float* __restrict__ fw2,
                                               const float* __restrict__ fb2,
                                               const int* __restrict__ ei,
                                               float* __restrict__ Mbuf) {
  int e = blockIdx.x * 256 + threadIdx.x;  // < EFWD
  int s = ei[e], d = ei[ETOT + e];
  const float* a = h2 + (size_t)s * 128;
  const float* b = h2 + (size_t)d * 128;
  float dot = 0.0f, na = 0.0f, nb = 0.0f;
  for (int j = 0; j < 128; ++j) {
    float av = a[j], bv = b[j];
    dot += av * bv; na += av * av; nb += bv * bv;
  }
  na = fmaxf(sqrtf(na), 1e-6f);
  nb = fmaxf(sqrtf(nb), 1e-6f);
  float cosv = dot / (na * nb);
  float ax0 = co[s * 4], ay0 = co[s * 4 + 1], ax1 = co[s * 4 + 2], ay1 = co[s * 4 + 3];
  float bx0 = co[d * 4], by0 = co[d * 4 + 1], bx1 = co[d * 4 + 2], by1 = co[d * 4 + 3];
  float iw = fmaxf(fminf(ax1, bx1) - fmaxf(ax0, bx0), 0.0f);
  float ih = fmaxf(fminf(ay1, by1) - fmaxf(ay0, by0), 0.0f);
  float inter = iw * ih;
  float iou = inter / ((ax1 - ax0) * (ay1 - ay0) + (bx1 - bx0) * (by1 - by0) - inter + 1e-6f);
  float score = fb2[0];
  for (int j = 0; j < 16; ++j)
    score += fmaxf(fb1[j] + cosv * fw1[j] + iou * fw1[16 + j], 0.0f) * fw2[j];
  int g = s >> 7;
  int row = s & 127;
  int col = (d & 127) - TT;
  Mbuf[g * 65 * 65 + row * 65 + col] = expf(score * LAMV);
}

// ---------------------------------------------------------------------------
// Sinkhorn: 8 iterations of row then column normalization on 65x65 in LDS.
// ---------------------------------------------------------------------------
__global__ void __launch_bounds__(256) k_sinkhorn(const float* __restrict__ Mbuf,
                                                  float* __restrict__ out) {
  __shared__ float Mm[65 * 65];  // 16.5KB
  __shared__ float ssum[65];
  int tid = threadIdx.x;
  int g = blockIdx.x;
  float slack = expf(-0.2f * LAMV);
  for (int i = tid; i < 65 * 65; i += 256) {
    int r = i / 65, cc = i - r * 65;
    Mm[i] = (r == 64 || cc == 64) ? slack : Mbuf[g * 65 * 65 + i];
  }
  __syncthreads();
  for (int it = 0; it < 8; ++it) {
    if (tid < 65) {
      float s = 0.0f;
      for (int cc = 0; cc < 65; ++cc) s += Mm[tid * 65 + cc];
      ssum[tid] = s;
    }
    __syncthreads();
    for (int i = tid; i < 65 * 65; i += 256) Mm[i] /= ssum[i / 65];
    __syncthreads();
    if (tid < 65) {
      float s = 0.0f;
      for (int r = 0; r < 65; ++r) s += Mm[r * 65 + tid];
      ssum[tid] = s;
    }
    __syncthreads();
    for (int i = tid; i < 65 * 65; i += 256) Mm[i] /= ssum[i - (i / 65) * 65];
    __syncthreads();
  }
  for (int i = tid; i < 64 * 64; i += 256) {
    int r = i >> 6, cc = i & 63;
    out[(size_t)g * 4096 + i] = Mm[r * 65 + cc];
  }
}

// ---------------------------------------------------------------------------
extern "C" void kernel_launch(void* const* d_in, const int* in_sizes, int n_in,
                              void* d_out, int out_size, void* d_ws, size_t ws_size,
                              hipStream_t stream) {
  (void)in_sizes; (void)n_in; (void)out_size; (void)ws_size;
  const float* x       = (const float*)d_in[0];
  const float* coords  = (const float*)d_in[1];
  const float* co      = (const float*)d_in[2];
  const float* pea     = (const float*)d_in[3];
  /* d_in[4] = ground_truth (unused in forward) */
  const float* conv1_w = (const float*)d_in[5];
  const float* conv1_b = (const float*)d_in[6];
  const float* conv2_w = (const float*)d_in[7];
  const float* conv2_b = (const float*)d_in[8];
  const float* fc_w    = (const float*)d_in[9];
  const float* fc_b    = (const float*)d_in[10];
  const float* app_w1  = (const float*)d_in[11];
  const float* app_b1  = (const float*)d_in[12];
  const float* app_w2  = (const float*)d_in[13];
  const float* app_b2  = (const float*)d_in[14];
  const float* geo_w1  = (const float*)d_in[15];
  const float* geo_b1  = (const float*)d_in[16];
  const float* geo_w2  = (const float*)d_in[17];
  const float* geo_b2  = (const float*)d_in[18];
  const float* aff_w1  = (const float*)d_in[19];
  const float* aff_b1  = (const float*)d_in[20];
  const float* aff_w2  = (const float*)d_in[21];
  const float* aff_b2  = (const float*)d_in[22];
  const float* pos_w1  = (const float*)d_in[23];
  const float* pos_b1  = (const float*)d_in[24];
  const float* pos_w2  = (const float*)d_in[25];
  const float* pos_b2  = (const float*)d_in[26];
  const float* msg_w   = (const float*)d_in[27];
  const float* msg_b   = (const float*)d_in[28];
  const float* upd_w   = (const float*)d_in[29];
  const float* upd_b   = (const float*)d_in[30];
  const float* fin_w1  = (const float*)d_in[31];
  const float* fin_b1  = (const float*)d_in[32];
  const float* fin_w2  = (const float*)d_in[33];
  const float* fin_b2  = (const float*)d_in[34];
  const int*   ei      = (const int*)d_in[35];
  float* out = (float*)d_out;

  char* wsB = (char*)d_ws;
  size_t off = 0;
  auto take = [&](size_t bytes) -> void* {
    void* p = wsB + off;
    off += (bytes + 255) & ~(size_t)255;
    return p;
  };
  _Float16* c2wr16 = (_Float16*)take((size_t)64 * 288 * 2);
  _Float16* fcw16  = (_Float16*)take((size_t)64 * 128 * 2);    // fragment-packed
  _Float16* appw16 = (_Float16*)take((size_t)256 * 64 * 2);    // fragment-packed
  _Float16* msgw16 = (_Float16*)take((size_t)160 * 128 * 2);   // fragment-packed
  _Float16* updw16 = (_Float16*)take((size_t)256 * 128 * 2);   // fragment-packed
  _Float16* c1g    = (_Float16*)take((size_t)NNODE * 32768 * 2);   // 64MB
  _Float16* pool16 = (_Float16*)take((size_t)NNODE * 64 * 2);
  _Float16* h16    = (_Float16*)take((size_t)NNODE * 128 * 2);
  float*    x1f    = (float*)take((size_t)ETOT * 4);
  _Float16* ef16   = (_Float16*)take((size_t)ETOT * 32 * 2);
  float*    msgf   = (float*)take((size_t)ETOT * 128 * 4);          // 33.5MB
  _Float16* agg16  = (_Float16*)take((size_t)NNODE * 128 * 2);
  float*    h2f    = (float*)take((size_t)NNODE * 128 * 4);
  float*    Mbuf   = (float*)take((size_t)BB * 65 * 65 * 4);

  // Weight prep: f16 conversion into WMMA-fragment-major layout.
  k_prep_packB<<<(64 * 128 + 255) / 256, 256, 0, stream>>>(fc_w, fcw16, 64, 128);
  k_prep_packB<<<(256 * 64 + 255) / 256, 256, 0, stream>>>(app_w1, appw16, 256, 64);
  k_prep_packB<<<(160 * 128 + 255) / 256, 256, 0, stream>>>(msg_w, msgw16, 160, 128);
  k_prep_packB<<<(256 * 128 + 255) / 256, 256, 0, stream>>>(upd_w, updw16, 256, 128);
  k_prep_c2<<<(64 * 288 + 255) / 256, 256, 0, stream>>>(conv2_w, c2wr16);

  // CNN backbone
  k_conv1<<<NNODE, 256, 0, stream>>>(x, conv1_w, conv1_b, c1g);
  k_conv2<<<NNODE, 256, 0, stream>>>(c1g, c2wr16, conv2_b, pool16);
  k_fc<<<NNODE / 64, 256, 0, stream>>>(pool16, fcw16, fc_b, h16);

  // Edge pipeline
  k_app<<<ETOT / 64, 256, 0, stream>>>(h16, appw16, app_b1, app_w2, app_b2, ei, x1f);
  k_edgefeat<<<ETOT / 256, 256, 0, stream>>>(coords, pea, x1f,
                                             geo_w1, geo_b1, geo_w2, geo_b2,
                                             aff_w1, aff_b1, aff_w2, aff_b2,
                                             pos_w1, pos_b1, pos_w2, pos_b2,
                                             ei, ef16);
  k_msg<<<ETOT / 64, 256, 0, stream>>>(h16, ef16, msgw16, msg_b, ei, msgf);
  k_agg<<<(NNODE * 128) / 256, 256, 0, stream>>>(msgf, agg16);
  k_upd<<<NNODE / 64, 256, 0, stream>>>(h16, agg16, updw16, upd_b, h2f);

  // Scoring + Sinkhorn
  k_score<<<EFWD / 256, 256, 0, stream>>>(h2f, co, fin_w1, fin_b1, fin_w2, fin_b2, ei, Mbuf);
  k_sinkhorn<<<BB, 256, 0, stream>>>(Mbuf, out);
}
